// SpatialSelfAttention_26869315404040
// MI455X (gfx1250) — matrix operand order repaired and checked by reading, not compiled, not run on hardware
//
#include <hip/hip_runtime.h>
#include <stdint.h>

#define NB   2
#define CCH  512
#define HWN  4096
#define NGRP 32
#define CPG  16
#define EPS  1e-6f
#define WMAT (CCH * CCH)   // 262144 elements per weight matrix

typedef __attribute__((ext_vector_type(16))) __bf16 v16bf;
typedef __attribute__((ext_vector_type(8)))  float  v8f;

union BfBits { unsigned short s; __bf16 h; };
union F8     { uint4 u4; __bf16 h[8]; };
union F4b    { uint2 u2; __bf16 h[4]; };

__device__ inline unsigned short f2bf_bits(float f) {
  union { float f; uint32_t u; } c; c.f = f;
  uint32_t u = c.u;
  u += 0x7FFFu + ((u >> 16) & 1u);           // round-to-nearest-even
  return (unsigned short)(u >> 16);
}
__device__ inline __bf16 f2bfh(float f) { BfBits b; b.s = f2bf_bits(f); return b.h; }

// A/B fragment fill: two contiguous 8-element (16B) bf16 chunks.
__device__ inline void frag_b16(v16bf& a, const __bf16* p0, const __bf16* p1) {
  F8 x, y;
  x.u4 = *(const uint4*)p0;
  y.u4 = *(const uint4*)p1;
#pragma unroll
  for (int i = 0; i < 8; ++i) { a[i] = x.h[i]; a[i + 8] = y.h[i]; }
}

__device__ inline v8f wmma_bf16(const v16bf& a, const v16bf& b, const v8f& c) {
  return __builtin_amdgcn_wmma_f32_16x16x32_bf16(false, a, false, b, (short)0, c,
                                                 false, false);
}

// ---------------- Kernel 0: weights fp32 -> bf16 (one shot, 2 MB) -----------
__global__ __launch_bounds__(256) void wcvt_kernel(
    const float* __restrict__ wq, const float* __restrict__ wk,
    const float* __restrict__ wv, const float* __restrict__ wp,
    __bf16* __restrict__ Wb) {
  int e0 = (blockIdx.x * 256 + threadIdx.x) * 4;
  int m = e0 >> 18;                                    // which matrix
  const float* src = m == 0 ? wq : m == 1 ? wk : m == 2 ? wv : wp;
  float4 f = *(const float4*)(src + (e0 & (WMAT - 1)));
  F4b o;
  o.h[0] = f2bfh(f.x); o.h[1] = f2bfh(f.y);
  o.h[2] = f2bfh(f.z); o.h[3] = f2bfh(f.w);
  *(uint2*)(Wb + e0) = o.u2;
}

// ---------------- Kernel 1: GroupNorm + transpose to [b, n, c] bf16 ----------
__global__ __launch_bounds__(256) void gn_kernel(
    const float* __restrict__ x, const float* __restrict__ gscale,
    const float* __restrict__ gbias, __bf16* __restrict__ T) {
  int b = blockIdx.x / NGRP, g = blockIdx.x % NGRP;
  int c0 = g * CPG;
  const float* xg = x + ((size_t)b * CCH + c0) * HWN;
  __shared__ float r1[256], r2[256];
  __shared__ float tile[CPG][257];
  float s = 0.f, s2 = 0.f;
  for (int i = threadIdx.x; i < CPG * HWN; i += 256) {
    float v = xg[i];
    s += v; s2 += v * v;
  }
  r1[threadIdx.x] = s; r2[threadIdx.x] = s2;
  __syncthreads();
  for (int off = 128; off > 0; off >>= 1) {
    if ((int)threadIdx.x < off) {
      r1[threadIdx.x] += r1[threadIdx.x + off];
      r2[threadIdx.x] += r2[threadIdx.x + off];
    }
    __syncthreads();
  }
  float mean = r1[0] * (1.0f / (CPG * HWN));
  float var  = r2[0] * (1.0f / (CPG * HWN)) - mean * mean;
  float inv  = rsqrtf(var + EPS);
  float al[CPG], bl[CPG];
#pragma unroll
  for (int cc = 0; cc < CPG; ++cc) {
    float a = inv * gscale[c0 + cc];
    al[cc] = a;
    bl[cc] = gbias[c0 + cc] - mean * a;
  }
  __syncthreads();
  for (int nb = 0; nb < HWN; nb += 256) {
#pragma unroll
    for (int cc = 0; cc < CPG; ++cc)
      tile[cc][threadIdx.x] = xg[(size_t)cc * HWN + nb + threadIdx.x];
    __syncthreads();
    int n = nb + threadIdx.x;
    F8 p0, p1;
#pragma unroll
    for (int cc = 0; cc < 8; ++cc)
      p0.h[cc] = f2bfh(tile[cc][threadIdx.x] * al[cc] + bl[cc]);
#pragma unroll
    for (int cc = 0; cc < 8; ++cc)
      p1.h[cc] = f2bfh(tile[cc + 8][threadIdx.x] * al[cc + 8] + bl[cc + 8]);
    uint4* dst = (uint4*)(T + ((size_t)b * HWN + n) * CCH + c0);
    dst[0] = p0.u4; dst[1] = p1.u4;
    __syncthreads();
  }
}

// ---------------- Kernel 2: fused QKV GEMMs ---------------------------------
// D[o, n] = W[o, :] . T[n, :]  (A = bf16 weights, B = staged LDS tile)
// Q, K stored [b, n, c]; V stored [b, c, n].
__global__ __launch_bounds__(256) void qkv_kernel(
    const __bf16* __restrict__ T, const __bf16* __restrict__ Wb,
    const float* __restrict__ bq, const float* __restrict__ bk,
    const float* __restrict__ bv,
    __bf16* __restrict__ Q, __bf16* __restrict__ Kt,
    __bf16* __restrict__ Vt) {
  const int NT = HWN / 32;
  int t = blockIdx.x;
  int which = t % 3; t /= 3;
  int nt = t % NT;  int b = t / NT;
  int n0 = nt * 32;
  const __bf16* W  = Wb + (size_t)which * WMAT;
  const float*  Bv = which == 0 ? bq : (which == 1 ? bk : bv);

  __shared__ __bf16 Bs[32][520];                       // 33 KB, padded rows
  for (int i = threadIdx.x; i < 32 * 64; i += 256) {
    int row = i >> 6, c8 = i & 63;
    *(uint4*)&Bs[row][c8 * 8] =
        *(const uint4*)(T + ((size_t)b * HWN + n0 + row) * CCH + c8 * 8);
  }
  __syncthreads();

  int lane = threadIdx.x & 31, wave = threadIdx.x >> 5;
  int hl = lane >> 4, l16 = lane & 15;

  for (int ot = wave; ot < CCH / 16; ot += 8) {
    int o0 = ot * 16;
    const __bf16* wr = W + (size_t)(o0 + l16) * CCH;
    float bo[8];
#pragma unroll
    for (int v = 0; v < 8; ++v) bo[v] = Bv[o0 + v + 8 * hl];
    v8f acc[2] = {};
    for (int k0 = 0; k0 < CCH; k0 += 32) {
      v16bf a;
      frag_b16(a, wr + k0 + 16 * hl, wr + k0 + 16 * hl + 8);
#pragma unroll
      for (int j = 0; j < 2; ++j) {
        v16bf bb;
        const __bf16* p = &Bs[16 * j + l16][k0 + 16 * hl];
        frag_b16(bb, p, p + 8);
        acc[j] = wmma_bf16(a, bb, acc[j]);
      }
    }
    if (which != 2) {
      // [n, c] layout: the 8 accumulator rows are contiguous in c -> b128 store
      __bf16* D = (which == 0 ? Q : Kt) + (size_t)b * HWN * CCH;
#pragma unroll
      for (int j = 0; j < 2; ++j) {
        F8 pk;
#pragma unroll
        for (int v = 0; v < 8; ++v) pk.h[v] = f2bfh(acc[j][v] + bo[v]);
        int n = n0 + 16 * j + l16;
        *(uint4*)(D + (size_t)n * CCH + o0 + 8 * hl) = pk.u4;
      }
    } else {
      __bf16* D = Vt + (size_t)b * CCH * HWN;
#pragma unroll
      for (int j = 0; j < 2; ++j)
#pragma unroll
        for (int v = 0; v < 8; ++v)
          D[(size_t)(o0 + v + 8 * hl) * HWN + n0 + 16 * j + l16] =
              f2bfh(acc[j][v] + bo[v]);
    }
  }
}

// ---------------- Kernel 3: flash attention ---------------------------------
__global__ __launch_bounds__(256) void attn_kernel(
    const __bf16* __restrict__ Q, const __bf16* __restrict__ Kt,
    const __bf16* __restrict__ Vt, __bf16* __restrict__ O) {
  int b  = blockIdx.x / (HWN / 16);
  int n0 = (blockIdx.x % (HWN / 16)) * 16;

  __shared__ __bf16 Qs[16][520];
  __shared__ __bf16 Ps[16][136];
  __shared__ float m_s[16], l_s[16], sc_s[16];
  __shared__ float wmax[8][16], wsum[8][16];

  for (int i = threadIdx.x; i < 16 * 64; i += 256) {
    int row = i >> 6, c8 = i & 63;
    *(uint4*)&Qs[row][c8 * 8] =
        *(const uint4*)(Q + ((size_t)b * HWN + n0 + row) * CCH + c8 * 8);
  }
  if (threadIdx.x < 16) { m_s[threadIdx.x] = -3.0e38f; l_s[threadIdx.x] = 0.f; }
  __syncthreads();

  int lane = threadIdx.x & 31, wave = threadIdx.x >> 5;
  int hl = lane >> 4, l16 = lane & 15;
  int cw0 = wave * 64;

  v8f accO[4] = {};
  const float qscale = 0.044194173824159216f;  // 1/sqrt(512)

  for (int m0 = 0; m0 < HWN; m0 += 128) {
    // prefetch next K tile rows into L2 while this tile computes
    if (m0 + 128 < HWN)
      __builtin_prefetch(Kt + ((size_t)b * HWN + m0 + 128 + wave * 16 + l16) * CCH,
                         0, 1);
    // ---- S = Q . K^T for this wave's 16 m-columns
    int mc = m0 + wave * 16 + l16;
    const __bf16* krow = Kt + ((size_t)b * HWN + mc) * CCH;
    v8f s = {};
    for (int k0 = 0; k0 < CCH; k0 += 32) {
      v16bf a, bb;
      const __bf16* qp = &Qs[l16][k0 + 16 * hl];
      frag_b16(a, qp, qp + 8);
      const __bf16* kp = krow + k0 + 16 * hl;
      frag_b16(bb, kp, kp + 8);
      s = wmma_bf16(a, bb, s);
    }
    // ---- per-wave row max (reduce across 16-lane half)
    float rm[8];
#pragma unroll
    for (int v = 0; v < 8; ++v) {
      float z = s[v] * qscale;
      s[v] = z;
#pragma unroll
      for (int off = 8; off > 0; off >>= 1) z = fmaxf(z, __shfl_xor(z, off, 32));
      rm[v] = z;
    }
    if (l16 == 0) {
#pragma unroll
      for (int v = 0; v < 8; ++v) wmax[wave][v + 8 * hl] = rm[v];
    }
    __syncthreads();
    if (threadIdx.x < 16) {
      int r = threadIdx.x;
      float mx = m_s[r];
#pragma unroll
      for (int w = 0; w < 8; ++w) mx = fmaxf(mx, wmax[w][r]);
      sc_s[r] = __expf(m_s[r] - mx);
      m_s[r]  = mx;
    }
    __syncthreads();
    // ---- P = exp(S - m), partial row sums, write bf16 probs to LDS
    float nm[8], rs[8];
#pragma unroll
    for (int v = 0; v < 8; ++v) nm[v] = m_s[v + 8 * hl];
#pragma unroll
    for (int v = 0; v < 8; ++v) {
      float p = __expf(s[v] - nm[v]);
      float z = p;
#pragma unroll
      for (int off = 8; off > 0; off >>= 1) z += __shfl_xor(z, off, 32);
      rs[v] = z;
      Ps[v + 8 * hl][wave * 16 + l16] = f2bfh(p);
    }
    if (l16 == 0) {
#pragma unroll
      for (int v = 0; v < 8; ++v) wsum[wave][v + 8 * hl] = rs[v];
    }
    // ---- rescale running O by exp(m_old - m_new)
    float f[8];
#pragma unroll
    for (int v = 0; v < 8; ++v) f[v] = sc_s[v + 8 * hl];
#pragma unroll
    for (int j = 0; j < 4; ++j)
#pragma unroll
      for (int v = 0; v < 8; ++v) accO[j][v] *= f[v];
    __syncthreads();
    if (threadIdx.x < 16) {
      int r = threadIdx.x;
      float acc = l_s[r] * sc_s[r];
#pragma unroll
      for (int w = 0; w < 8; ++w) acc += wsum[w][r];
      l_s[r] = acc;
    }
    // ---- O += P . V for this wave's c slice (V stored [c, n])
#pragma unroll
    for (int k0 = 0; k0 < 128; k0 += 32) {
      v16bf a;
      const __bf16* pp = &Ps[l16][k0 + 16 * hl];
      frag_b16(a, pp, pp + 8);
#pragma unroll
      for (int j = 0; j < 4; ++j) {
        v16bf bb;
        const __bf16* vp = Vt + ((size_t)b * CCH + cw0 + 16 * j + l16) * HWN +
                           m0 + k0 + 16 * hl;
        frag_b16(bb, vp, vp + 8);
        accO[j] = wmma_bf16(a, bb, accO[j]);
      }
    }
    __syncthreads();
  }
  // ---- normalize, store O [b, n, c] bf16
  float inv[8];
#pragma unroll
  for (int v = 0; v < 8; ++v) inv[v] = 1.0f / l_s[v + 8 * hl];
#pragma unroll
  for (int j = 0; j < 4; ++j)
#pragma unroll
    for (int v = 0; v < 8; ++v) {
      int n = n0 + v + 8 * hl;
      int c = cw0 + 16 * j + l16;
      O[((size_t)b * HWN + n) * CCH + c] = f2bfh(accO[j][v] * inv[v]);
    }
}

// ---------------- Kernel 4: out-projection + bias + residual ----------------
__global__ __launch_bounds__(256) void proj_kernel(
    const __bf16* __restrict__ O, const __bf16* __restrict__ Wb,
    const float* __restrict__ bp, const float* __restrict__ x,
    float* __restrict__ out) {
  const int NT = HWN / 32;
  int t = blockIdx.x;
  int nt = t % NT, b = t / NT;
  int n0 = nt * 32;
  const __bf16* W = Wb + (size_t)3 * WMAT;
  __shared__ __bf16 Bs[32][520];
  for (int i = threadIdx.x; i < 32 * 64; i += 256) {
    int row = i >> 6, c8 = i & 63;
    *(uint4*)&Bs[row][c8 * 8] =
        *(const uint4*)(O + ((size_t)b * HWN + n0 + row) * CCH + c8 * 8);
  }
  __syncthreads();
  int lane = threadIdx.x & 31, wave = threadIdx.x >> 5;
  int hl = lane >> 4, l16 = lane & 15;
  for (int ot = wave; ot < CCH / 16; ot += 8) {
    int o0 = ot * 16;
    const __bf16* wr = W + (size_t)(o0 + l16) * CCH;
    v8f acc[2] = {};
    for (int k0 = 0; k0 < CCH; k0 += 32) {
      v16bf a;
      frag_b16(a, wr + k0 + 16 * hl, wr + k0 + 16 * hl + 8);
#pragma unroll
      for (int j = 0; j < 2; ++j) {
        v16bf bb;
        const __bf16* p = &Bs[16 * j + l16][k0 + 16 * hl];
        frag_b16(bb, p, p + 8);
        acc[j] = wmma_bf16(a, bb, acc[j]);
      }
    }
    float bo[8];
#pragma unroll
    for (int v = 0; v < 8; ++v) bo[v] = bp[o0 + v + 8 * hl];
#pragma unroll
    for (int j = 0; j < 2; ++j)
#pragma unroll
      for (int v = 0; v < 8; ++v) {
        int o = o0 + v + 8 * hl;
        int n = n0 + 16 * j + l16;
        size_t idx = ((size_t)b * CCH + o) * HWN + n;
        out[idx] = x[idx] + bo[v] + acc[j][v];
      }
  }
}

extern "C" void kernel_launch(void* const* d_in, const int* in_sizes, int n_in,
                              void* d_out, int out_size, void* d_ws,
                              size_t ws_size, hipStream_t stream) {
  const float* x  = (const float*)d_in[0];
  const float* gs = (const float*)d_in[1];
  const float* gb = (const float*)d_in[2];
  const float* wq = (const float*)d_in[3];
  const float* bq = (const float*)d_in[4];
  const float* wk = (const float*)d_in[5];
  const float* bk = (const float*)d_in[6];
  const float* wv = (const float*)d_in[7];
  const float* bv = (const float*)d_in[8];
  const float* wp = (const float*)d_in[9];
  const float* bp = (const float*)d_in[10];
  float* out = (float*)d_out;

  char* ws = (char*)d_ws;
  const size_t SZ = (size_t)NB * HWN * CCH * sizeof(__bf16);  // 8 MB
  __bf16* T  = (__bf16*)(ws);
  __bf16* Qb = (__bf16*)(ws + 1 * SZ);
  __bf16* Kb = (__bf16*)(ws + 2 * SZ);
  __bf16* Vb = (__bf16*)(ws + 3 * SZ);
  __bf16* Ob = (__bf16*)(ws + 4 * SZ);
  __bf16* Wc = (__bf16*)(ws + 5 * SZ);                        // 2 MB

  wcvt_kernel<<<(4 * WMAT / 4) / 256, 256, 0, stream>>>(wq, wk, wv, wp, Wc);
  gn_kernel<<<NB * NGRP, 256, 0, stream>>>(x, gs, gb, T);
  qkv_kernel<<<NB * (HWN / 32) * 3, 256, 0, stream>>>(T, Wc, bq, bk, bv,
                                                      Qb, Kb, Vb);
  attn_kernel<<<NB * (HWN / 16), 256, 0, stream>>>(Qb, Kb, Vb, Ob);
  proj_kernel<<<NB * (HWN / 32), 256, 0, stream>>>(Ob, Wc, bp, x, out);
}
